// Temporal_Mamba_8143257993725
// MI455X (gfx1250) — compile-verified, matrix operand order checked by reading, not run on hardware
//
#include <hip/hip_runtime.h>

typedef __attribute__((ext_vector_type(16))) __bf16 v16bf;
typedef __attribute__((ext_vector_type(8)))  float  v8f;

namespace {
constexpr int Cc   = 96;      // channels
constexpr int Tt   = 16;      // timesteps
constexpr int HWc  = 3136;    // 56*56
constexpr int NH   = 3;       // heads
constexpr int DH   = 32;      // head dim
constexpr int TILE = 16;      // batch rows per block
constexpr int NTHR = 192;     // 6 wave32: one per (head, coltile)
constexpr int NPAIR = TILE * Cc;     // 1536 (row, channel) pairs
constexpr int PPT   = NPAIR / NTHR;  // 8 pairs per thread
}

__global__ __launch_bounds__(NTHR) void slstm_fused_kernel(
    const float* __restrict__ x,       // (B, C, T, H, W)
    const float* __restrict__ conv_w,  // (C, 4)
    const float* __restrict__ conv_b,  // (C,)
    const float* __restrict__ Wi,      // (NH, DH, DH)
    const float* __restrict__ Wf,
    const float* __restrict__ Wz,
    const float* __restrict__ Wo,
    const float* __restrict__ Rm,      // (4, NH, DH, DH)
    const float* __restrict__ bias,    // (4, NH, DH)
    const float* __restrict__ gnw,     // (C,)
    float* __restrict__ out)           // (B, C, T, H, W)
{
    // -------- LDS (~40 KB) --------
    __shared__ float  s_ring[4][TILE][Cc];  // f32 x taps, async-loaded ring
    __shared__ __bf16 s_xch[TILE][Cc];      // silu(conv) (A for gi/gf)
    __shared__ __bf16 s_xsb[TILE][Cc];      // xs(t)      (A for gz/go)
    __shared__ __bf16 s_h  [TILE][Cc];      // hidden state (A for recurrence)
    __shared__ float  s_ys [TILE][Cc];      // h(t) for GroupNorm
    __shared__ float  s_mu [TILE * NH];
    __shared__ float  s_rs [TILE * NH];

    const int tidx  = threadIdx.x;
    const int wave  = tidx >> 5;           // 0..5
    const int lane  = tidx & 31;
    const int ncol  = lane & 15;           // A row / B col / D col within tile
    const int kb    = (lane >> 4) << 3;    // K sub-block: 0 or 8
    const int rbase = kb;                  // D row base:  0 or 8
    const int head  = wave >> 1;           // wave owns (head, coltile), all 4 gates
    const int ct    = wave & 1;
    const int cbase = head * DH;
    const int bc    = ct * 16 + ncol;      // column inside 32-wide head matrix
    const int col   = cbase + bc;          // global channel column this lane owns

    // ---- preload B fragments (W_g, R_g) + biases for all 4 gates in registers ----
    v16bf Wfrag[4], Rfrag[4];
    float bval[4];
    const float* Wmats[4] = {Wi, Wf, Wz, Wo};
    #pragma unroll
    for (int g = 0; g < 4; ++g) {
        const float* Wp = Wmats[g] + head * DH * DH;        // W[head][k][e]
        const float* Rp = Rm + (g * NH + head) * DH * DH;   // R[g][head][k][e]
        #pragma unroll
        for (int j = 0; j < 8; ++j) {
            Wfrag[g][j]     = (__bf16)Wp[(kb + j)      * DH + bc];
            Wfrag[g][8 + j] = (__bf16)Wp[(16 + kb + j) * DH + bc];
            Rfrag[g][j]     = (__bf16)Rp[(kb + j)      * DH + bc];
            Rfrag[g][8 + j] = (__bf16)Rp[(16 + kb + j) * DH + bc];
        }
        bval[g] = bias[(g * NH + head) * DH + bc];
    }

    // ---- sLSTM state lives in registers, D-fragment layout (8 rows per lane) ----
    float cS[8], nS[8], mS[8];
    #pragma unroll
    for (int r = 0; r < 8; ++r) { cS[r] = 0.f; nS[r] = 0.f; mS[r] = 0.f; }
    #pragma unroll
    for (int q = 0; q < PPT; ++q) {
        int p = tidx + q * NTHR;
        s_h[p & (TILE - 1)][p >> 4] = (__bf16)0.f;
    }
    __syncthreads();

    // ================= time loop =================
    for (int t = 0; t < Tt; ++t) {
        const int slot = t & 3;

        // ---- phase 0: async copy tap t (16x96 f32) global -> LDS ring ----
        #pragma unroll
        for (int q = 0; q < PPT; ++q) {
            int p   = tidx + q * NTHR;
            int row = p & (TILE - 1);
            int cg  = p >> 4;
            int bf  = blockIdx.x * TILE + row;
            int bb  = bf / HWc;
            int hw  = bf - bb * HWc;
            const float* gp = x + ((size_t)(bb * Cc + cg) * Tt + t) * HWc + hw;
            unsigned int       la = (unsigned int)(size_t)&s_ring[slot][row][cg];
            unsigned long long ga = (unsigned long long)(size_t)gp;
            asm volatile("global_load_async_to_lds_b32 %0, %1, off"
                         :: "v"(la), "v"(ga) : "memory");
        }
        asm volatile("s_wait_asynccnt 0x0" ::: "memory");
        __syncthreads();

        // ---- phase 1: depthwise causal conv (K=4) + SiLU from f32 ring ----
        #pragma unroll
        for (int q = 0; q < PPT; ++q) {
            int p   = tidx + q * NTHR;
            int row = p & (TILE - 1);
            int cg  = p >> 4;
            float acc = conv_b[cg];
            #pragma unroll
            for (int k = 0; k < 4; ++k) {
                int tt = t - 3 + k;
                if (tt >= 0)
                    acc = fmaf(s_ring[tt & 3][row][cg], conv_w[cg * 4 + k], acc);
            }
            float sg = __builtin_amdgcn_rcpf(1.f + __expf(-acc));
            s_xch[row][cg] = (__bf16)(acc * sg);
            s_xsb[row][cg] = (__bf16)s_ring[slot][row][cg];
        }
        __syncthreads();

        // ---- phase 2a: build A fragments (read h BEFORE anyone rewrites it) ----
        v16bf Axc, Axs, Ah;
        {
            int m = ncol;
            #pragma unroll
            for (int j = 0; j < 8; ++j) {
                Axc[j]     = s_xch[m][cbase + kb + j];
                Axc[8 + j] = s_xch[m][cbase + 16 + kb + j];
                Axs[j]     = s_xsb[m][cbase + kb + j];
                Axs[8 + j] = s_xsb[m][cbase + 16 + kb + j];
                Ah[j]      = s_h [m][cbase + kb + j];
                Ah[8 + j]  = s_h [m][cbase + 16 + kb + j];
            }
        }
        __syncthreads();

        // ---- phase 2b: 8 WMMAs (4 gates x {x-proj, recurrence}) ----
        v8f D[4];
        #pragma unroll
        for (int g = 0; g < 4; ++g) {
            v8f ci;
            #pragma unroll
            for (int r = 0; r < 8; ++r) ci[r] = bval[g];
            const v16bf& Ax = (g < 2) ? Axc : Axs;   // i,f use conv path; z,o raw x
            D[g] = __builtin_amdgcn_wmma_f32_16x16x32_bf16(
                       false, Ax, false, Wfrag[g], (short)0, ci, false, false);
            D[g] = __builtin_amdgcn_wmma_f32_16x16x32_bf16(
                       false, Ah, false, Rfrag[g], (short)0, D[g], false, false);
        }

        // ---- phase 2c: stabilized sLSTM update fully in registers ----
        #pragma unroll
        for (int r = 0; r < 8; ++r) {
            int row = rbase + r;
            float iraw = D[0][r], fraw = D[1][r], zraw = D[2][r], oraw = D[3][r];
            // log_sigmoid(f) = min(f,0) - log(1 + exp(-|f|))
            float lsf = fminf(fraw, 0.f) - __logf(1.f + __expf(-fabsf(fraw)));
            float lfm = mS[r] + lsf;
            float mn  = fmaxf(iraw, lfm);
            float ig  = __expf(iraw - mn);
            float fg  = __expf(lfm - mn);
            float e2  = __expf(2.f * zraw);                       // tanh(z)
            float th  = 1.f - 2.f * __builtin_amdgcn_rcpf(e2 + 1.f);
            float cn  = fg * cS[r] + ig * th;
            float nn  = fg * nS[r] + ig;
            float so  = __builtin_amdgcn_rcpf(1.f + __expf(-oraw));
            float hn  = so * cn * __builtin_amdgcn_rcpf(nn);
            cS[r] = cn; nS[r] = nn; mS[r] = mn;
            s_h [row][col] = (__bf16)hn;
            s_ys[row][col] = hn;
        }
        __syncthreads();

        // ---- phase 3: GroupNorm stats (per row, per head over DH=32) ----
        if (tidx < TILE * NH) {
            int row = tidx / NH;
            int hd  = tidx % NH;
            float s = 0.f, s2 = 0.f;
            #pragma unroll
            for (int d = 0; d < DH; ++d) {
                float v = s_ys[row][hd * DH + d];
                s  += v;
                s2 += v * v;
            }
            float mu  = s * (1.f / DH);
            float var = s2 * (1.f / DH) - mu * mu;
            s_mu[tidx] = mu;
            s_rs[tidx] = __builtin_amdgcn_rcpf(__builtin_amdgcn_sqrtf(var + 1e-5f));
        }
        __syncthreads();

        // ---- phase 4: normalize + transposed coalesced store ----
        #pragma unroll
        for (int q = 0; q < PPT; ++q) {
            int p   = tidx + q * NTHR;
            int row = p & (TILE - 1);
            int cg  = p >> 4;
            int hd  = cg >> 5;
            int bf  = blockIdx.x * TILE + row;
            int bb  = bf / HWc;
            int hw  = bf - bb * HWc;
            float yn = (s_ys[row][cg] - s_mu[row * NH + hd])
                       * s_rs[row * NH + hd] * gnw[cg];
            out[((size_t)(bb * Cc + cg) * Tt + t) * HWc + hw] = yn;
        }
        __syncthreads();
    }
}

extern "C" void kernel_launch(void* const* d_in, const int* in_sizes, int n_in,
                              void* d_out, int out_size, void* d_ws, size_t ws_size,
                              hipStream_t stream) {
    const float* x      = (const float*)d_in[0];
    const float* conv_w = (const float*)d_in[1];
    const float* conv_b = (const float*)d_in[2];
    const float* Wi     = (const float*)d_in[3];
    const float* Wf     = (const float*)d_in[4];
    const float* Wz     = (const float*)d_in[5];
    const float* Wo     = (const float*)d_in[6];
    const float* Rm     = (const float*)d_in[7];
    const float* bias   = (const float*)d_in[8];
    const float* gnw    = (const float*)d_in[9];
    float* out          = (float*)d_out;

    // Bf = 2*56*56 = 6272 rows, 16 rows per block -> 392 blocks
    slstm_fused_kernel<<<392, NTHR, 0, stream>>>(
        x, conv_w, conv_b, Wi, Wf, Wz, Wo, Rm, bias, gnw, out);
}
